// MultiHeadAttention_66297115181647
// MI455X (gfx1250) — matrix operand order repaired and checked by reading, not compiled, not run on hardware
//
#include <hip/hip_runtime.h>
#include <hip/hip_bf16.h>

// ---------------- problem constants ----------------
#define D_MODEL   1024
#define NUM_HEADS 16
#define D_K       64
#define BATCH     2
#define SEQ       2048
#define M_TOT     (BATCH * SEQ)   // 4096

// ---------------- WMMA fragment types ----------------
typedef __attribute__((ext_vector_type(16))) __bf16 v16bf;
typedef __attribute__((ext_vector_type(8)))  float  v8f;
// TDM descriptor groups
typedef __attribute__((ext_vector_type(4))) unsigned int u32x4;
typedef __attribute__((ext_vector_type(8))) int          i32x8;
typedef __attribute__((ext_vector_type(4))) int          i32x4;

union FragBF { v16bf v; uint4 u[2]; };

__device__ __forceinline__ v8f zero8() {
  v8f z = {0.f, 0.f, 0.f, 0.f, 0.f, 0.f, 0.f, 0.f};
  return z;
}

// fp32 -> bf16 bits, round to nearest even
__device__ __forceinline__ unsigned f2bf_bits(float f) {
  unsigned u = __float_as_uint(f);
  u += 0x7FFFu + ((u >> 16) & 1u);
  return u >> 16;
}
__device__ __forceinline__ __bf16 f2bf(float f) {
  unsigned short h = (unsigned short)f2bf_bits(f);
  return __builtin_bit_cast(__bf16, h);
}

// ---------------- Tensor Data Mover: 2D tile global->LDS ----------------
// Copies tile1 rows x tile0 elements (2-byte elements) from a row-major
// tensor with row stride `stride0` (elements) into a dense LDS tile.
// D# packing per CDNA5 ISA ch.8 (group0 128b, group1 256b; groups 2/3 = 0).
// This toolchain exposes the 6-arg builtin (extra int32x8 group, zero-filled).
__device__ __forceinline__ void tdm_load_2d(unsigned lds_off, const void* gptr,
                                            unsigned tile0, unsigned tile1,
                                            unsigned stride0) {
  const unsigned long long ga = (unsigned long long)(uintptr_t)gptr;
  u32x4 g0;
  g0[0] = 1u;                                            // count=1, user mode
  g0[1] = lds_off;                                       // lds_addr (bytes)
  g0[2] = (unsigned)ga;                                  // global_addr[31:0]
  g0[3] = (unsigned)((ga >> 32) & 0x01FFFFFFu) | (2u << 30);  // addr[56:32] | type=2
  i32x8 g1;
  g1[0] = (int)(1u << 16);                               // data_size = 2 bytes
  g1[1] = (int)((tile0 & 0xFFFFu) << 16);                // tensor_dim0[15:0]
  g1[2] = (int)(((tile0 >> 16) & 0xFFFFu) |              // tensor_dim0[31:16]
                ((tile1 & 0xFFFFu) << 16));              // tensor_dim1[15:0]
  g1[3] = (int)(((tile1 >> 16) & 0xFFFFu) |              // tensor_dim1[31:16]
                ((tile0 & 0xFFFFu) << 16));              // tile_dim0
  g1[4] = (int)(tile1 & 0xFFFFu);                        // tile_dim1 (tile_dim2=0)
  g1[5] = (int)stride0;                                  // tensor_dim0_stride[31:0]
  g1[6] = 0;                                             // stride0[47:32] | dim1_stride lo
  g1[7] = 0;
  i32x4 z4 = {0, 0, 0, 0};
  i32x8 z8 = {0, 0, 0, 0, 0, 0, 0, 0};
  __builtin_amdgcn_tensor_load_to_lds(g0, g1, z4, z4, z8, 0);
}

// ---------------- fp32 -> bf16 conversion ----------------
__global__ void cvt_f32_bf16(const float* __restrict__ in, __bf16* __restrict__ out, int n) {
  int i = blockIdx.x * blockDim.x + threadIdx.x;
  if (i < n) out[i] = f2bf(in[i]);
}

// ---------------- WMMA GEMM: C[M,N] = A[M,K] * B[N,K]^T ----------------
// MODE 0: out bf16 row-major [M,N]
// MODE 1: out bf16 scattered as vT[b, h, d, s]   (V projection)
// MODE 2: out fp32 row-major [M,N]               (final projection)
// 256 threads = 8 waves (4x2), wave tile 32x32, block tile 128x64.
template <int MODE>
__global__ void __launch_bounds__(256)
gemm_nt(const __bf16* __restrict__ A, const __bf16* __restrict__ B,
        void* __restrict__ C, int M, int N, int K) {
  const int lane = threadIdx.x & 31;
  const int wave = threadIdx.x >> 5;
  const int wm   = wave >> 1;
  const int wn   = wave & 1;
  const int m0   = blockIdx.y * 128 + wm * 32;
  const int n0   = blockIdx.x * 64  + wn * 32;
  const int lr   = lane & 15;
  const int hi   = lane >> 4;

  v8f acc[2][2];
#pragma unroll
  for (int i = 0; i < 2; ++i)
#pragma unroll
    for (int j = 0; j < 2; ++j) acc[i][j] = zero8();

  for (int kk = 0; kk < K; kk += 32) {
    FragBF a[2], b[2];
#pragma unroll
    for (int i = 0; i < 2; ++i) {
      const __bf16* p = A + (size_t)(m0 + i * 16 + lr) * K + kk + hi * 8;
      a[i].u[0] = *reinterpret_cast<const uint4*>(p);
      a[i].u[1] = *reinterpret_cast<const uint4*>(p + 16);
    }
#pragma unroll
    for (int j = 0; j < 2; ++j) {
      const __bf16* p = B + (size_t)(n0 + j * 16 + lr) * K + kk + hi * 8;
      b[j].u[0] = *reinterpret_cast<const uint4*>(p);
      b[j].u[1] = *reinterpret_cast<const uint4*>(p + 16);
    }
    __builtin_prefetch(A + (size_t)(m0 + lr) * K + kk + 32, 0, 1);
#pragma unroll
    for (int i = 0; i < 2; ++i)
#pragma unroll
      for (int j = 0; j < 2; ++j)
        acc[i][j] = __builtin_amdgcn_wmma_f32_16x16x32_bf16(
            false, a[i].v, false, b[j].v, (short)0, acc[i][j], false, false);
  }

#pragma unroll
  for (int i = 0; i < 2; ++i)
#pragma unroll
    for (int j = 0; j < 2; ++j)
#pragma unroll
      for (int r = 0; r < 8; ++r) {
        const int row = m0 + i * 16 + r + hi * 8;
        const int col = n0 + j * 16 + lr;
        const float val = acc[i][j][r];
        if (MODE == 2) {
          ((float*)C)[(size_t)row * N + col] = val;
        } else if (MODE == 0) {
          ((__bf16*)C)[(size_t)row * N + col] = f2bf(val);
        } else {
          const int b_ = row >> 11;
          const int s_ = row & (SEQ - 1);
          const int h_ = col >> 6;
          const int d_ = col & (D_K - 1);
          ((__bf16*)C)[(((size_t)(b_ * NUM_HEADS + h_) * D_K + d_) * SEQ) + s_] =
              f2bf(val);
        }
      }
}

// ---------------- flash-style attention with TDM-fed LDS tiles ----------------
// qp, kp: bf16 [B*S, D_MODEL]; vT: bf16 [B, H, D_K, SEQ]; out: bf16 [B*S, D_MODEL].
// Block = 128 threads (4 waves) on one (b,h,64-row q-chunk); wave = 16 q rows.
// Wave 0 drives the Tensor Data Mover: double-buffered K (32x64) and V (64x32)
// tiles stream into LDS under TENSORcnt while all waves run WMMA.
__global__ void __launch_bounds__(128)
attn_kernel(const __bf16* __restrict__ qp, const __bf16* __restrict__ kp,
            const __bf16* __restrict__ vT, __bf16* __restrict__ out) {
  __shared__ __align__(16) __bf16 s_k[2][32][D_K];   // 2 x 4KB
  __shared__ __align__(16) __bf16 s_v[2][D_K][32];   // 2 x 4KB
  __shared__ __align__(16) float  s_sc[4][16][32];   // 8KB
  __shared__ __align__(16) __bf16 s_pr[4][16][32];   // 4KB

  const int lane = threadIdx.x & 31;
  const int wave = threadIdx.x >> 5;
  const int lr   = lane & 15;
  const int hi   = lane >> 4;

  const int qc = blockIdx.x & 31;
  const int h  = (blockIdx.x >> 5) & 15;
  const int b  = blockIdx.x >> 9;
  const int qbase = qc * 64 + wave * 16;

  const __bf16* kbase = kp + (size_t)(b * SEQ) * D_MODEL + h * D_K;
  const __bf16* vbase = vT + (size_t)(b * NUM_HEADS + h) * D_K * SEQ;

  // Q fragments: 16x64 as two 16x32 A-frags (loaded once from global)
  FragBF aq[2];
#pragma unroll
  for (int t = 0; t < 2; ++t) {
    const __bf16* p =
        qp + (size_t)(b * SEQ + qbase + lr) * D_MODEL + h * D_K + t * 32 + hi * 8;
    aq[t].u[0] = *reinterpret_cast<const uint4*>(p);
    aq[t].u[1] = *reinterpret_cast<const uint4*>(p + 16);
  }

  v8f o[4];
#pragma unroll
  for (int t = 0; t < 4; ++t) o[t] = zero8();
  float m_i = -1e30f, l_i = 0.f;

  // prologue: TDM for tile 0 into buffer 0
  if (wave == 0) {
    tdm_load_2d((unsigned)(uintptr_t)&s_k[0][0][0], kbase, D_K, 32, D_MODEL);
    tdm_load_2d((unsigned)(uintptr_t)&s_v[0][0][0], vbase, 32, D_K, SEQ);
  }

  for (int kv = 0; kv < SEQ; kv += 32) {
    const int buf = (kv >> 5) & 1;
    if (wave == 0) {
      if (kv + 32 < SEQ) {  // issue next tile, then retire the current pair
        tdm_load_2d((unsigned)(uintptr_t)&s_k[buf ^ 1][0][0],
                    kbase + (size_t)(kv + 32) * D_MODEL, D_K, 32, D_MODEL);
        tdm_load_2d((unsigned)(uintptr_t)&s_v[buf ^ 1][0][0],
                    vbase + (kv + 32), 32, D_K, SEQ);
        __builtin_amdgcn_s_wait_tensorcnt(2);
      } else {
        __builtin_amdgcn_s_wait_tensorcnt(0);
      }
    }
    __syncthreads();  // tile `buf` visible to all 4 waves

    // ---- scores: 16 x 32, K-dim = 64 (2 WMMA steps), B-frags from LDS ----
    v8f s[2];
#pragma unroll
    for (int j = 0; j < 2; ++j) {
      s[j] = zero8();
#pragma unroll
      for (int t = 0; t < 2; ++t) {
        FragBF bk;
        const __bf16* p = &s_k[buf][j * 16 + lr][t * 32 + hi * 8];
        bk.u[0] = *reinterpret_cast<const uint4*>(p);
        bk.u[1] = *reinterpret_cast<const uint4*>(p + 16);
        s[j] = __builtin_amdgcn_wmma_f32_16x16x32_bf16(
            false, aq[t].v, false, bk.v, (short)0, s[j], false, false);
      }
    }

    // ---- stage scaled scores to LDS (C layout -> [row][col]) ----
#pragma unroll
    for (int j = 0; j < 2; ++j)
#pragma unroll
      for (int r = 0; r < 8; ++r)
        s_sc[wave][r + hi * 8][j * 16 + lr] = s[j][r] * 0.125f;  // 1/sqrt(64)

    asm volatile("s_wait_dscnt 0" ::: "memory");

    // ---- full-wave online softmax: lane handles row lr, columns hi*16..+15 ----
    {
      const float* row = &s_sc[wave][lr][hi * 16];
      float tmax = -1e30f;
#pragma unroll
      for (int c = 0; c < 16; ++c) tmax = fmaxf(tmax, row[c]);
      tmax = fmaxf(tmax, __shfl_xor(tmax, 16, 32));  // combine column halves
      const float newm = fmaxf(m_i, tmax);
      const float alpha = __expf(m_i - newm);
      float psum = 0.f;
      unsigned* pr32 = (unsigned*)&s_pr[wave][lr][hi * 16];
#pragma unroll
      for (int c = 0; c < 16; c += 2) {  // packed 2 x bf16 stores
        const float p0 = __expf(row[c] - newm);
        const float p1 = __expf(row[c + 1] - newm);
        pr32[c >> 1] = f2bf_bits(p0) | (f2bf_bits(p1) << 16);
        psum += p0 + p1;
      }
      psum += __shfl_xor(psum, 16, 32);
      l_i = l_i * alpha + psum;  // state replicated in both lane halves
      m_i = newm;

      // rescale running output accumulators by alpha[row]
      float al[8];
#pragma unroll
      for (int r = 0; r < 8; ++r) al[r] = __shfl(alpha, r + hi * 8, 32);
#pragma unroll
      for (int t = 0; t < 4; ++t)
#pragma unroll
        for (int r = 0; r < 8; ++r) o[t][r] *= al[r];
    }
    asm volatile("s_wait_dscnt 0" ::: "memory");

    // ---- probs (16x32) as A-frag from LDS ----
    FragBF ap;
    {
      const __bf16* pp = &s_pr[wave][lr][hi * 8];
      ap.u[0] = *reinterpret_cast<const uint4*>(pp);
      ap.u[1] = *reinterpret_cast<const uint4*>(pp + 16);
    }

    // ---- PV: out[16 x 64] += P[16x32] @ V[32x64], B-frags from LDS ----
#pragma unroll
    for (int t = 0; t < 4; ++t) {
      FragBF bv;
      const __bf16* p = &s_v[buf][t * 16 + lr][hi * 8];
      bv.u[0] = *reinterpret_cast<const uint4*>(p);
      bv.u[1] = *reinterpret_cast<const uint4*>(p + 16);
      o[t] = __builtin_amdgcn_wmma_f32_16x16x32_bf16(
          false, ap.v, false, bv.v, (short)0, o[t], false, false);
    }
    __syncthreads();  // all waves done with tile `buf` before TDM reuses it
  }

  // ---- normalize by l_i and store (re-concatenate heads) ----
  const float inv = 1.f / l_i;
  float il[8];
#pragma unroll
  for (int r = 0; r < 8; ++r) il[r] = __shfl(inv, r + hi * 8, 32);
#pragma unroll
  for (int t = 0; t < 4; ++t)
#pragma unroll
    for (int r = 0; r < 8; ++r) {
      const int row = qbase + r + hi * 8;
      const int col = h * D_K + t * 16 + lr;
      out[(size_t)(b * SEQ + row) * D_MODEL + col] = f2bf(o[t][r] * il[r]);
    }
}

// ---------------- host-side orchestration ----------------
extern "C" void kernel_launch(void* const* d_in, const int* in_sizes, int n_in,
                              void* d_out, int out_size, void* d_ws, size_t ws_size,
                              hipStream_t stream) {
  const float* Q  = (const float*)d_in[0];
  const float* K  = (const float*)d_in[1];
  const float* V  = (const float*)d_in[2];
  const float* Wq = (const float*)d_in[3];
  const float* Wk = (const float*)d_in[4];
  const float* Wv = (const float*)d_in[5];
  const float* Wo = (const float*)d_in[6];

  char* ws = (char*)d_ws;
  size_t off = 0;
  auto alloc = [&](size_t bytes) {
    char* p = ws + off;
    off += (bytes + 255) & ~(size_t)255;
    return p;
  };

  const size_t NX = (size_t)M_TOT * D_MODEL;
  const size_t NW = (size_t)D_MODEL * D_MODEL;

  __bf16* Qb   = (__bf16*)alloc(NX * 2);
  __bf16* Kb   = (__bf16*)alloc(NX * 2);
  __bf16* Vb   = (__bf16*)alloc(NX * 2);
  __bf16* Wqb  = (__bf16*)alloc(NW * 2);
  __bf16* Wkb  = (__bf16*)alloc(NW * 2);
  __bf16* Wvb  = (__bf16*)alloc(NW * 2);
  __bf16* Wob  = (__bf16*)alloc(NW * 2);
  __bf16* qpr  = (__bf16*)alloc(NX * 2);
  __bf16* kpr  = (__bf16*)alloc(NX * 2);
  __bf16* vT   = (__bf16*)alloc(NX * 2);
  __bf16* aout = (__bf16*)alloc(NX * 2);

  const int CT = 256;
  cvt_f32_bf16<<<(int)((NX + CT - 1) / CT), CT, 0, stream>>>(Q, Qb, (int)NX);
  cvt_f32_bf16<<<(int)((NX + CT - 1) / CT), CT, 0, stream>>>(K, Kb, (int)NX);
  cvt_f32_bf16<<<(int)((NX + CT - 1) / CT), CT, 0, stream>>>(V, Vb, (int)NX);
  cvt_f32_bf16<<<(int)((NW + CT - 1) / CT), CT, 0, stream>>>(Wq, Wqb, (int)NW);
  cvt_f32_bf16<<<(int)((NW + CT - 1) / CT), CT, 0, stream>>>(Wk, Wkb, (int)NW);
  cvt_f32_bf16<<<(int)((NW + CT - 1) / CT), CT, 0, stream>>>(Wv, Wvb, (int)NW);
  cvt_f32_bf16<<<(int)((NW + CT - 1) / CT), CT, 0, stream>>>(Wo, Wob, (int)NW);

  dim3 gg(D_MODEL / 64, M_TOT / 128);  // (16, 32)
  gemm_nt<0><<<gg, 256, 0, stream>>>(Qb, Wqb, qpr, M_TOT, D_MODEL, D_MODEL);
  gemm_nt<0><<<gg, 256, 0, stream>>>(Kb, Wkb, kpr, M_TOT, D_MODEL, D_MODEL);
  gemm_nt<1><<<gg, 256, 0, stream>>>(Vb, Wvb, vT,  M_TOT, D_MODEL, D_MODEL);

  attn_kernel<<<BATCH * NUM_HEADS * (SEQ / 64), 128, 0, stream>>>(qpr, kpr, vT, aout);

  gemm_nt<2><<<gg, 256, 0, stream>>>(aout, Wob, d_out, M_TOT, D_MODEL, D_MODEL);
}
